// GenePhenAIv2_0_6511170421639
// MI455X (gfx1250) — matrix operand order repaired and checked by reading, not compiled, not run on hardware
//
#include <hip/hip_runtime.h>
#include <math.h>
#include <float.h>

#define N_NODES 50000
#define N_EDGES 800000
#define N_GROUPS 64
#define D 128
#define D_OUT 5000
#define EPS_BN 1e-5f

typedef __attribute__((ext_vector_type(16))) _Float16 v16h;
typedef __attribute__((ext_vector_type(8)))  float    v8f;

// ---------------------------------------------------------------- swizzle maps
// A-fragment (16x32 f16 per WMMA): lane = r + 16*hi holds row r; element j of the
// v16h maps to k_local = (j&7) + (j>>3)*16 + hi*8.  Inverse (store side):
__device__ __forceinline__ int swz_idx_a(int r16, int k) {   // k in [0,128)
    int kk = k >> 5, kl = k & 31;
    int hi = (kl >> 3) & 1;
    int j  = (kl & 7) + ((kl >> 4) << 3);
    int lane = r16 + (hi << 4);
    return ((kk << 5) + lane) * 16 + j;      // halves within one 16-row tile (2048)
}
// B-fragment (32x16 f16 per WMMA): lane = n + 16*((k_local>>3)&1), element j -> (k_local&7) + (k_local>>4)*8.
__device__ __forceinline__ int swz_idx_b(int k, int n) {     // k,n in [0,128)
    int wv = n >> 4, nn = n & 15;
    int kk = k >> 5, kl = k & 31;
    int lane = nn + (((kl >> 3) & 1) << 4);
    int j = (kl & 7) + ((kl >> 4) << 3);
    return (((wv << 2) + kk) * 32 + lane) * 16 + j;          // total 16384 halves
}

// ---------------------------------------------------------------- utilities
__global__ void fill_f32(float* __restrict__ p, float v, int n) {
    int i = blockIdx.x * blockDim.x + threadIdx.x;
    if (i < n) p[i] = v;
}

__device__ inline float atomicMaxF(float* addr, float val) {
    int* ai = (int*)addr;
    int old = __float_as_int(*addr);
    while (__int_as_float(old) < val) {
        int assumed = old;
        old = atomicCAS(ai, assumed, __float_as_int(val));
        if (old == assumed) break;
    }
    return __int_as_float(old);
}

// ------------------------------------------------------------ degree / norm
__global__ void deg_accum(const int* __restrict__ dst, float* __restrict__ deg, int ne) {
    int i = blockIdx.x * blockDim.x + threadIdx.x;
    if (i < ne) atomicAdd(&deg[dst[i]], 1.0f);
}

__global__ void deg_to_dis(float* __restrict__ deg, int n) {
    int i = blockIdx.x * blockDim.x + threadIdx.x;
    if (i < n) {
        float d = deg[i];
        d = d < 1.0f ? 1.0f : d;
        deg[i] = rsqrtf(d);   // in place: deg -> dis
    }
}

// ------------------------------------------------------- operand pre-swizzling
__global__ void w_to_swz(const float* __restrict__ W, _Float16* __restrict__ Wswz) {
    int e = blockIdx.x * blockDim.x + threadIdx.x;   // e < 128*128
    if (e >= D * D) return;
    int k = e >> 7, n = e & (D - 1);
    Wswz[swz_idx_b(k, n)] = (_Float16)W[e];
}

__global__ void x_to_swz(const float* __restrict__ X, _Float16* __restrict__ Aswz) {
    long i = (long)blockIdx.x * blockDim.x + threadIdx.x;
    if (i >= (long)N_NODES * D) return;
    int r = (int)(i >> 7), c = (int)(i & (D - 1));
    Aswz[(long)(r >> 4) * 2048 + swz_idx_a(r & 15, c)] = (_Float16)X[i];
}

// --------------------------------------------------------------- WMMA GEMM
// Fragment-swizzled, LDS-free: wave w of each 256-thread block owns one 16-row
// tile and computes all 8 column tiles (32 chained wmma per wave).
__global__ __launch_bounds__(256) void gemm_swz(const _Float16* __restrict__ Aswz,
                                                const _Float16* __restrict__ Wswz,
                                                float* __restrict__ C,
                                                int ntiles) {
    const int t    = threadIdx.x;
    const int lane = t & 31;
    const int wv   = t >> 5;
    const int tile = blockIdx.x * 8 + wv;
    if (tile >= ntiles) return;                       // wave-uniform exit

    const v16h* __restrict__ ap = (const v16h*)Aswz + (long)tile * 128; // 2048 halves
    const v16h* __restrict__ bp = (const v16h*)Wswz;

    const int m  = lane & 15;
    const int hi = lane >> 4;

    v8f acc[8] = {};
#pragma unroll
    for (int kk = 0; kk < 4; ++kk) {
        v16h a = ap[kk * 32 + lane];                  // 2x global_load_b128
#pragma unroll
        for (int w2 = 0; w2 < 8; ++w2) {
            v16h b = bp[((w2 << 2) + kk) * 32 + lane];
            acc[w2] = __builtin_amdgcn_wmma_f32_16x16x32_f16(false, a, false, b,
                                                             (short)0, acc[w2],
                                                             false, false);
        }
    }
    const long rowBase = (long)tile * 16;
#pragma unroll
    for (int w2 = 0; w2 < 8; ++w2)
#pragma unroll
        for (int r = 0; r < 8; ++r)
            C[(rowBase + hi * 8 + r) * D + w2 * 16 + m] = acc[w2][r];
}

// -------------------------------------------------- edge scatter (1 wave/edge)
__global__ __launch_bounds__(256) void scatter_add(const float* __restrict__ hw,
                                                   const int* __restrict__ src,
                                                   const int* __restrict__ dst,
                                                   const float* __restrict__ dis,
                                                   float* __restrict__ agg,
                                                   int etotal) {
    int wid  = (blockIdx.x * blockDim.x + threadIdx.x) >> 5;  // one edge per wave
    int lane = threadIdx.x & 31;
    if (wid >= etotal) return;
    int s, d;
    if (wid < N_EDGES) { s = src[wid]; d = dst[wid]; }
    else               { s = d = wid - N_EDGES; }              // self loops
    float nrm = dis[s] * dis[d];
    const float4* hp = reinterpret_cast<const float4*>(hw + (long)s * D);
    float4 v = hp[lane];
    float* op = agg + (long)d * D + lane * 4;
    atomicAdd(op + 0, v.x * nrm);
    atomicAdd(op + 1, v.y * nrm);
    atomicAdd(op + 2, v.z * nrm);
    atomicAdd(op + 3, v.w * nrm);
}

// ------------------------------------------------------------------ batchnorm
__global__ __launch_bounds__(128) void bn_stats(const float* __restrict__ agg,
                                                const float* __restrict__ bias,
                                                float* __restrict__ sums,
                                                float* __restrict__ sqs,
                                                int rows_per_block) {
    int c = threadIdx.x;
    long r0 = (long)blockIdx.x * rows_per_block;
    long r1 = r0 + rows_per_block;
    if (r1 > N_NODES) r1 = N_NODES;
    float b = bias[c];
    float s = 0.f, q = 0.f;
    for (long r = r0; r < r1; ++r) {
        float v = agg[r * D + c] + b;
        s += v; q += v * v;
    }
    atomicAdd(&sums[c], s);
    atomicAdd(&sqs[c], q);
}

__global__ void bn_final(const float* __restrict__ sums, const float* __restrict__ sqs,
                         float* __restrict__ mean, float* __restrict__ inv) {
    int c = threadIdx.x;
    float m = sums[c] * (1.0f / N_NODES);
    float v = sqs[c] * (1.0f / N_NODES) - m * m;
    mean[c] = m;
    inv[c]  = rsqrtf(v + EPS_BN);
}

// BN + ReLU; writes f32 h AND the f16 fragment-swizzled copy for the next GEMM.
__global__ __launch_bounds__(256) void bn_apply_relu(const float* __restrict__ agg,
                                                     const float* __restrict__ bias,
                                                     const float* __restrict__ mean,
                                                     const float* __restrict__ inv,
                                                     const float* __restrict__ g,
                                                     const float* __restrict__ be,
                                                     float* __restrict__ out,
                                                     _Float16* __restrict__ outSwz) {
    long i = (long)blockIdx.x * blockDim.x + threadIdx.x;
    if (i >= (long)N_NODES * D) return;
    int c = (int)(i & (D - 1));
    float v = (agg[i] + bias[c] - mean[c]) * inv[c] * g[c] + be[c];
    v = v > 0.f ? v : 0.f;
    out[i] = v;
    int r = (int)(i >> 7);
    outSwz[(long)(r >> 4) * 2048 + swz_idx_a(r & 15, c)] = (_Float16)v;
}

// ------------------------------------------------------------------ gate MLP
__global__ __launch_bounds__(64) void gate_kernel(const float* __restrict__ h,
                                                  const float* __restrict__ gw1,
                                                  const float* __restrict__ gb1,
                                                  const float* __restrict__ gw2,
                                                  const float* __restrict__ gb2,
                                                  float* __restrict__ gate) {
    __shared__ float hrow[D];
    __shared__ float red[64];
    int t = threadIdx.x;
    long node = blockIdx.x;
    hrow[t]      = h[node * D + t];
    hrow[t + 64] = h[node * D + t + 64];
    __syncthreads();
    float acc = gb1[t];
    for (int k = 0; k < D; ++k) acc += hrow[k] * gw1[k * 64 + t];
    acc = acc > 0.f ? acc : 0.f;
    red[t] = acc * gw2[t];
    __syncthreads();
    for (int off = 32; off > 0; off >>= 1) {
        if (t < off) red[t] += red[t + off];
        __syncthreads();
    }
    if (t == 0) gate[node] = red[0] + gb2[0];
}

// --------------------------------------------------------- softmax pooling
__global__ void seg_max(const float* __restrict__ gate, const int* __restrict__ batch,
                        float* __restrict__ m, int n) {
    int i = blockIdx.x * blockDim.x + threadIdx.x;
    if (i < n) atomicMaxF(&m[batch[i]], gate[i]);
}

__global__ void seg_expsum(const float* __restrict__ gate, const int* __restrict__ batch,
                           const float* __restrict__ m, float* __restrict__ e,
                           float* __restrict__ s, int n) {
    int i = blockIdx.x * blockDim.x + threadIdx.x;
    if (i < n) {
        int g = batch[i];
        float v = expf(gate[i] - m[g]);
        e[i] = v;
        atomicAdd(&s[g], v);
    }
}

__global__ __launch_bounds__(128) void pool_kernel(const float* __restrict__ h,
                                                   const float* __restrict__ e,
                                                   const float* __restrict__ s,
                                                   const int* __restrict__ batch,
                                                   float* __restrict__ pooled,
                                                   int nodes_per_block) {
    int c = threadIdx.x;
    long r0 = (long)blockIdx.x * nodes_per_block;
    long r1 = r0 + nodes_per_block;
    if (r1 > N_NODES) r1 = N_NODES;
    float acc = 0.f;
    int cur = -1;
    for (long r = r0; r < r1; ++r) {
        int g = batch[r];              // batch is sorted -> few group switches
        if (g != cur) {
            if (cur >= 0) atomicAdd(&pooled[cur * D + c], acc);
            acc = 0.f; cur = g;
        }
        acc += h[r * D + c] * (e[r] / s[g]);
    }
    if (cur >= 0) atomicAdd(&pooled[cur * D + c], acc);
}

// ----------------------------------------------------------------- classifier
__global__ __launch_bounds__(256) void classifier(const float* __restrict__ pooled,
                                                  const float* __restrict__ cw,
                                                  const float* __restrict__ cb,
                                                  float* __restrict__ out) {
    long i = (long)blockIdx.x * blockDim.x + threadIdx.x;
    if (i >= (long)N_GROUPS * D_OUT) return;
    int g = (int)(i / D_OUT);
    int o = (int)(i % D_OUT);
    float acc = cb[o];
    for (int k = 0; k < D; ++k)
        acc += pooled[g * D + k] * cw[(long)k * D_OUT + o];
    out[i] = acc;
}

// ================================================================= launcher
extern "C" void kernel_launch(void* const* d_in, const int* in_sizes, int n_in,
                              void* d_out, int out_size, void* d_ws, size_t ws_size,
                              hipStream_t stream) {
    (void)in_sizes; (void)n_in; (void)out_size; (void)ws_size;
    const float* x     = (const float*)d_in[0];
    const int*   ei    = (const int*)d_in[1];
    const int*   batch = (const int*)d_in[2];
    const float* w[4]  = {(const float*)d_in[3],  (const float*)d_in[7],
                          (const float*)d_in[11], (const float*)d_in[15]};
    const float* bb[4] = {(const float*)d_in[4],  (const float*)d_in[8],
                          (const float*)d_in[12], (const float*)d_in[16]};
    const float* gg[4] = {(const float*)d_in[5],  (const float*)d_in[9],
                          (const float*)d_in[13], (const float*)d_in[17]};
    const float* be[4] = {(const float*)d_in[6],  (const float*)d_in[10],
                          (const float*)d_in[14], (const float*)d_in[18]};
    const float* gw1 = (const float*)d_in[19];
    const float* gb1 = (const float*)d_in[20];
    const float* gw2 = (const float*)d_in[21];
    const float* gb2 = (const float*)d_in[22];
    const float* cw  = (const float*)d_in[23];
    const float* cb  = (const float*)d_in[24];
    float* out = (float*)d_out;

    const int* srcIdx = ei;            // edge_index[0]
    const int* dstIdx = ei + N_EDGES;  // edge_index[1]

    // -------- workspace layout --------
    float* ws = (float*)d_ws;
    size_t off = 0;
    float* dis    = ws + off; off += N_NODES;          // deg -> dis in place
    float* hw     = ws + off; off += (size_t)N_NODES * D;
    float* agg    = ws + off; off += (size_t)N_NODES * D;
    float* hbuf   = ws + off; off += (size_t)N_NODES * D;
    float* sums   = ws + off; off += D;
    float* sqs    = ws + off; off += D;
    float* meanv  = ws + off; off += D;
    float* invv   = ws + off; off += D;
    float* gate   = ws + off; off += N_NODES;
    float* eArr   = ws + off; off += N_NODES;
    float* mval   = ws + off; off += N_GROUPS;
    float* sval   = ws + off; off += N_GROUPS;
    float* pooled = ws + off; off += (size_t)N_GROUPS * D;
    _Float16* aswz = (_Float16*)(ws + off); off += (size_t)N_NODES * D / 2;
    _Float16* wswz = (_Float16*)(ws + off); off += (4 * D * D) / 2;

    const int ET = N_EDGES + N_NODES;            // edges incl. self loops
    const long NH = (long)N_NODES * D;
    const int ntiles = N_NODES / 16;             // 3125 (exact)

    // degree / symmetric norm
    fill_f32<<<(N_NODES + 255) / 256, 256, 0, stream>>>(dis, 1.0f, N_NODES);
    deg_accum<<<(N_EDGES + 255) / 256, 256, 0, stream>>>(dstIdx, dis, N_EDGES);
    deg_to_dis<<<(N_NODES + 255) / 256, 256, 0, stream>>>(dis, N_NODES);

    // pre-swizzle weights (f16 fragment order) and the layer-0 input
    for (int l = 0; l < 4; ++l)
        w_to_swz<<<(D * D + 255) / 256, 256, 0, stream>>>(w[l], wswz + l * D * D);
    x_to_swz<<<(int)((NH + 255) / 256), 256, 0, stream>>>(x, aswz);

    const int gemmBlocks = (ntiles + 7) / 8;     // 391
    const int bnBlocks = 500, bnRows = (N_NODES + bnBlocks - 1) / bnBlocks;

    for (int l = 0; l < 4; ++l) {
        gemm_swz<<<gemmBlocks, 256, 0, stream>>>(aswz, wswz + l * D * D, hw, ntiles);
        fill_f32<<<(int)((NH + 255) / 256), 256, 0, stream>>>(agg, 0.0f, (int)NH);
        scatter_add<<<(ET * 32 + 255) / 256, 256, 0, stream>>>(hw, srcIdx, dstIdx,
                                                               dis, agg, ET);
        fill_f32<<<1, 256, 0, stream>>>(sums, 0.0f, 2 * D);   // sums+sqs contiguous
        bn_stats<<<bnBlocks, 128, 0, stream>>>(agg, bb[l], sums, sqs, bnRows);
        bn_final<<<1, 128, 0, stream>>>(sums, sqs, meanv, invv);
        bn_apply_relu<<<(int)((NH + 255) / 256), 256, 0, stream>>>(agg, bb[l], meanv,
                                                                   invv, gg[l], be[l],
                                                                   hbuf, aswz);
    }

    // gating + softmax pooling
    gate_kernel<<<N_NODES, 64, 0, stream>>>(hbuf, gw1, gb1, gw2, gb2, gate);
    fill_f32<<<1, 64, 0, stream>>>(mval, -FLT_MAX, N_GROUPS);
    fill_f32<<<1, 64, 0, stream>>>(sval, 0.0f, N_GROUPS);
    seg_max<<<(N_NODES + 255) / 256, 256, 0, stream>>>(gate, batch, mval, N_NODES);
    seg_expsum<<<(N_NODES + 255) / 256, 256, 0, stream>>>(gate, batch, mval, eArr,
                                                          sval, N_NODES);
    fill_f32<<<(N_GROUPS * D + 255) / 256, 256, 0, stream>>>(pooled, 0.0f,
                                                             N_GROUPS * D);
    pool_kernel<<<(N_NODES + 127) / 128, 128, 0, stream>>>(hbuf, eArr, sval, batch,
                                                           pooled, 128);

    // classifier
    classifier<<<(N_GROUPS * D_OUT + 255) / 256, 256, 0, stream>>>(pooled, cw, cb, out);
}
